// AdvancedPermutationTreeLayerRNN_18227841204268
// MI455X (gfx1250) — compile-verified, matrix operand order checked by reading, not run on hardware
//
#include <hip/hip_runtime.h>
#include <hip/hip_bf16.h>
#include <math.h>

// ---------------------------------------------------------------------------
// Types
// ---------------------------------------------------------------------------
typedef __attribute__((ext_vector_type(16))) __bf16        bf16x16;
typedef __attribute__((ext_vector_type(8)))  float         f32x8;
typedef __attribute__((ext_vector_type(4)))  float         f32x4;
typedef __attribute__((ext_vector_type(4)))  unsigned int  u32x4;
typedef __attribute__((ext_vector_type(2)))  unsigned int  u32x2;

union FragAB { u32x4 u[2]; bf16x16 v; };
union Pack4  { unsigned short s[4]; u32x2 u; };

#define HDIM   128
#define KCHILD 4
#define WELEMS (HDIM * HDIM)   // 16384 per weight matrix

static __device__ __forceinline__ unsigned short f32_to_bf16(float f) {
  unsigned int u = __builtin_bit_cast(unsigned int, f);
  u += 0x7FFFu + ((u >> 16) & 1u);           // round-to-nearest-even
  return (unsigned short)(u >> 16);
}

static __device__ __forceinline__ float eluf(float x) {
  return x > 0.0f ? x : expm1f(x);
}

// ---------------------------------------------------------------------------
// Half-width GEMM: C[16x64](4 f32 frags) += A[16x128](bf16 LDS) @ W^T, output
// tiles tbase..tbase+3. 4 accumulator frags = 32 VGPRs keeps the wave under
// the 256-VGPR direct window (no s_set_vgpr_msb) and allows >=4 waves/SIMD.
// W bf16 row-major [128][128]; B[k][n] = W[n][k] -> contiguous row reads.
// ---------------------------------------------------------------------------
static __device__ __forceinline__ void gemm_half(
    const unsigned short* a,                  // LDS bf16 [16*128]
    const unsigned short* __restrict__ w,     // global bf16 [128*128]
    f32x8 c[4], int lr, int half, int tbase) {
  for (int kb = 0; kb < HDIM; kb += 32) {
    FragAB af;
    const u32x4* ap = (const u32x4*)(a + lr * HDIM + kb + half * 8);
    af.u[0] = ap[0];
    af.u[1] = ap[2];                          // +16 elems = +32B
#pragma unroll
    for (int tt = 0; tt < 4; ++tt) {
      FragAB bf;
      const u32x4* bp =
          (const u32x4*)(w + ((tbase + tt) * 16 + lr) * HDIM + kb + half * 8);
      bf.u[0] = bp[0];
      bf.u[1] = bp[2];
      c[tt] = __builtin_amdgcn_wmma_f32_16x16x32_bf16(
          false, af.v, false, bf.v, (short)0, c[tt], false, false);
    }
  }
}

static __device__ __forceinline__ void zeroC4(f32x8 c[4]) {
  f32x8 z = {};
#pragma unroll
  for (int t = 0; t < 4; ++t) c[t] = z;
}

// D half-frags -> LDS f32 [16][128]. lane(lr,half), vgpr r: row=half*8+r, col=t*16+lr
static __device__ __forceinline__ void storeD_f32_half(float* dst, const f32x8 c[4],
                                                       int lr, int half, int tbase) {
#pragma unroll
  for (int tt = 0; tt < 4; ++tt)
#pragma unroll
    for (int r = 0; r < 8; ++r)
      dst[(half * 8 + r) * HDIM + (tbase + tt) * 16 + lr] = c[tt][r];
}

// D half-frags -> LDS bf16 with tanh (RNN step) or elu(tanh(.)) (last step)
static __device__ __forceinline__ void storeD_bf16_act_half(unsigned short* dst,
                                                            const f32x8 c[4], int lr,
                                                            int half, int tbase,
                                                            bool applyElu) {
#pragma unroll
  for (int tt = 0; tt < 4; ++tt)
#pragma unroll
    for (int r = 0; r < 8; ++r) {
      float v = tanhf(c[tt][r]);
      if (applyElu) v = eluf(v);
      dst[(half * 8 + r) * HDIM + (tbase + tt) * 16 + lr] = f32_to_bf16(v);
    }
}

// Masked segment-sum of a 16x128 LDS tile into global acc[P][128] with
// run-length merging (pooling is sorted -> same-group rows are contiguous).
static __device__ __forceinline__ void accum_masked(const float* __restrict__ sel,
                                                    const int* __restrict__ grp,
                                                    const int* __restrict__ mask,
                                                    float* __restrict__ acc, int lane) {
#pragma unroll
  for (int j = 0; j < 4; ++j) {
    int col = lane + 32 * j;
    float run = 0.0f;
    int gp = grp[0];
    for (int r = 0; r < 16; ++r) {
      int g = grp[r];
      if (g != gp) {
        if (run != 0.0f) atomicAdd(acc + (size_t)gp * HDIM + col, run);
        run = 0.0f;
        gp = g;
      }
      if (mask[r]) run += sel[r * HDIM + col];
    }
    if (run != 0.0f) atomicAdd(acc + (size_t)gp * HDIM + col, run);
  }
}

// RNN branch: h_t = tanh(A_t@Wih^T + h_{t-1}@Whh^T); elem = elu(h_K)@Wf^T; seg-sum.
// H ping-pongs between two LDS buffers (a half-GEMM writes new-h columns while
// the other half still reads every column of old-h).
static __device__ __forceinline__ void rnn_branch(
    const unsigned short* A,                   // LDS bf16 [4][16*128]
    unsigned short* Hbuf0, unsigned short* Hbuf1,
    float* SEL,                                // LDS f32 [16*128]
    const unsigned short* __restrict__ Wih,
    const unsigned short* __restrict__ Whh,
    const unsigned short* __restrict__ Wf,
    const int* __restrict__ grp, const int* __restrict__ mask,
    float* __restrict__ acc, int lr, int half, int lane) {
#pragma unroll 1
  for (int hb = 0; hb < 2; ++hb) {             // step 0: h0 = 0 -> Wih only
    f32x8 c[4];
    zeroC4(c);
    gemm_half(A, Wih, c, lr, half, hb * 4);
    storeD_bf16_act_half(Hbuf0, c, lr, half, hb * 4, false);
  }
  __syncthreads();
  unsigned short* Hprev = Hbuf0;
  unsigned short* Hnext = Hbuf1;
#pragma unroll 1
  for (int t = 1; t < KCHILD; ++t) {
#pragma unroll 1
    for (int hb = 0; hb < 2; ++hb) {
      f32x8 c[4];
      zeroC4(c);
      gemm_half(A + t * (16 * HDIM), Wih, c, lr, half, hb * 4);
      gemm_half(Hprev, Whh, c, lr, half, hb * 4);
      storeD_bf16_act_half(Hnext, c, lr, half, hb * 4, t == KCHILD - 1);
    }
    __syncthreads();
    unsigned short* tmp = Hprev; Hprev = Hnext; Hnext = tmp;
  }
#pragma unroll 1
  for (int hb = 0; hb < 2; ++hb) {             // Hprev holds elu(h_K)
    f32x8 c[4];
    zeroC4(c);
    gemm_half(Hprev, Wf, c, lr, half, hb * 4);
    storeD_f32_half(SEL, c, lr, half, hb * 4);
  }
  __syncthreads();
  accum_masked(SEL, grp, mask, acc, lane);
  __syncthreads();
}

// ---------------------------------------------------------------------------
// Kernel: init workspace accumulators / type_after
// ---------------------------------------------------------------------------
__global__ void init_ws_kernel(float* __restrict__ acc, long long nAcc,
                               int* __restrict__ ta, int P) {
  long long i = (long long)blockIdx.x * blockDim.x + threadIdx.x;
  long long stride = (long long)gridDim.x * blockDim.x;
  for (long long j = i; j < nAcc; j += stride) acc[j] = 0.0f;
  for (long long j = i; j < P; j += stride) ta[j] = -1;
}

// ---------------------------------------------------------------------------
// Kernel: convert 8 weight matrices fp32 -> bf16 (row-major, kept L2-hot)
// ---------------------------------------------------------------------------
__global__ void conv_weights_kernel(const float* __restrict__ w0, const float* __restrict__ w1,
                                    const float* __restrict__ w2, const float* __restrict__ w3,
                                    const float* __restrict__ w4, const float* __restrict__ w5,
                                    const float* __restrict__ w6, const float* __restrict__ w7,
                                    unsigned short* __restrict__ wbf) {
  int i = blockIdx.x * blockDim.x + threadIdx.x;
  if (i >= WELEMS) return;
  const float* ws[8] = {w0, w1, w2, w3, w4, w5, w6, w7};
#pragma unroll
  for (int k = 0; k < 8; ++k) wbf[k * WELEMS + i] = f32_to_bf16(ws[k][i]);
}

// ---------------------------------------------------------------------------
// Kernel: type_after = segment_max(type_mask, pooling)
// ---------------------------------------------------------------------------
__global__ void type_after_kernel(const int* __restrict__ pooling,
                                  const int* __restrict__ type_mask,
                                  int* __restrict__ ta, int M) {
  int m = blockIdx.x * blockDim.x + threadIdx.x;
  if (m < M) atomicMax(&ta[pooling[m]], type_mask[m]);
}

// ---------------------------------------------------------------------------
// Main kernel: one wave (32 lanes) per 16-row m-tile.
// Gathers children -> bf16 LDS, runs P linear + Z/S RNNs via bf16 WMMA,
// seg-sums into accP/accZ/accS; type-0 passthrough into acc0 during gather.
// wbf layout: 0=W_p 1=W_pf 2=zWih 3=zWhh 4=zWf 5=sWih 6=sWhh 7=sWf
// ---------------------------------------------------------------------------
__global__ __launch_bounds__(32)
__attribute__((amdgpu_waves_per_eu(4)))
void tree_rnn_main(
    const float* __restrict__ x, const int* __restrict__ initial_map,
    const int* __restrict__ order_matrix, const int* __restrict__ pooling,
    const int* __restrict__ type_mask, const unsigned short* __restrict__ wbf,
    float* __restrict__ accP, float* __restrict__ accZ,
    float* __restrict__ accS, float* __restrict__ acc0, int M) {
  const int lane = threadIdx.x;
  const int lr = lane & 15, half = lane >> 4;
  const int base = blockIdx.x * 16;
  if (base >= M) return;

  __shared__ unsigned short A[KCHILD][16 * HDIM];   // 16 KB bf16 child tiles
  __shared__ unsigned short Hb0[16 * HDIM];         // 4 KB RNN hidden ping
  __shared__ unsigned short Hb1[16 * HDIM];         // 4 KB RNN hidden pong
  __shared__ float SEL[16 * HDIM];                  // 8 KB branch result (f32)
  __shared__ int grp[16], rowidx[KCHILD][16];
  __shared__ int mP[16], mZ[16], mS[16], m0v[16];

  if (lane < 16) {
    int m = base + lane;
    bool valid = m < M;
    if (!valid) m = M - 1;
    int g = pooling[m];
    int tm = type_mask[m];
    int allE = 1;
#pragma unroll
    for (int k = 0; k < KCHILD; ++k) {
      int om = order_matrix[(size_t)k * M + m];     // -1 == empty slot
      rowidx[k][lane] = (om < 0) ? -1 : initial_map[om];
      if (om >= 0) allE = 0;
    }
    grp[lane] = g;
    mP[lane]  = valid && (tm == 1);
    mZ[lane]  = valid && !allE && (tm == 2 || tm == 4);
    mS[lane]  = valid && !allE && (tm == 3 || tm == 5);
    m0v[lane] = valid && (tm == 0);
  }
  __syncthreads();

  // Gather children as bf16 tiles; fold type-0 passthrough seg-sum in here.
  for (int idx = 0; idx < KCHILD * 16; ++idx) {
    int k = idx >> 4, r = idx & 15;
    int ri = rowidx[k][r];
    f32x4 v = {};
    if (ri >= 0) v = *(const f32x4*)(x + (size_t)ri * HDIM + lane * 4);
    Pack4 p;
#pragma unroll
    for (int i = 0; i < 4; ++i) p.s[i] = f32_to_bf16(v[i]);
    *(u32x2*)&A[k][r * HDIM + lane * 4] = p.u;
    if (k == 0 && m0v[r]) {
      float* dst = acc0 + (size_t)grp[r] * HDIM + lane * 4;
#pragma unroll
      for (int i = 0; i < 4; ++i) atomicAdd(dst + i, v[i]);
    }
  }
  __syncthreads();

  // ---- P branch: gathered[0] @ W_p^T -> masked seg-sum
#pragma unroll 1
  for (int hb = 0; hb < 2; ++hb) {
    f32x8 c[4];
    zeroC4(c);
    gemm_half(&A[0][0], wbf + 0 * WELEMS, c, lr, half, hb * 4);
    storeD_f32_half(SEL, c, lr, half, hb * 4);
  }
  __syncthreads();
  accum_masked(SEL, grp, mP, accP, lane);
  __syncthreads();

  // ---- Z / S RNN branches
  rnn_branch(&A[0][0], Hb0, Hb1, SEL, wbf + 2 * WELEMS, wbf + 3 * WELEMS,
             wbf + 4 * WELEMS, grp, mZ, accZ, lr, half, lane);
  rnn_branch(&A[0][0], Hb0, Hb1, SEL, wbf + 5 * WELEMS, wbf + 6 * WELEMS,
             wbf + 7 * WELEMS, grp, mS, accS, lr, half, lane);
}

// ---------------------------------------------------------------------------
// Finalize: per 16-group tile, out = select(type_after):
//   P: accP @ W_pf^T   Z: elu(accZ)   S: elu(accS)   0: acc0   else 0
// ---------------------------------------------------------------------------
__global__ __launch_bounds__(32) void finalize_kernel(
    const float* __restrict__ accP, const float* __restrict__ accZ,
    const float* __restrict__ accS, const float* __restrict__ acc0,
    const int* __restrict__ typeAfter, const unsigned short* __restrict__ wpf,
    float* __restrict__ out, int P) {
  const int lane = threadIdx.x;
  const int lr = lane & 15, half = lane >> 4;
  const int base = blockIdx.x * 16;
  if (base >= P) return;

  __shared__ unsigned short Abuf[16 * HDIM];
  __shared__ float SEL[16 * HDIM];

  for (int r = 0; r < 16; ++r) {
    int p = base + r;
    if (p >= P) p = P - 1;
    f32x4 v = *(const f32x4*)(accP + (size_t)p * HDIM + lane * 4);
    Pack4 pk;
#pragma unroll
    for (int i = 0; i < 4; ++i) pk.s[i] = f32_to_bf16(v[i]);
    *(u32x2*)&Abuf[r * HDIM + lane * 4] = pk.u;
  }
  __syncthreads();

#pragma unroll 1
  for (int hb = 0; hb < 2; ++hb) {
    f32x8 c[4];
    zeroC4(c);
    gemm_half(Abuf, wpf, c, lr, half, hb * 4);
    storeD_f32_half(SEL, c, lr, half, hb * 4);
  }
  __syncthreads();

  for (int r = 0; r < 16; ++r) {
    int p = base + r;
    if (p >= P) break;
    int ta = typeAfter[p];
#pragma unroll
    for (int j = 0; j < 4; ++j) {
      int col = lane + 32 * j;
      float o;
      if (ta == 1)                    o = SEL[r * HDIM + col];
      else if (ta == 2 || ta == 4)    o = eluf(accZ[(size_t)p * HDIM + col]);
      else if (ta == 3 || ta == 5)    o = eluf(accS[(size_t)p * HDIM + col]);
      else if (ta == 0)               o = acc0[(size_t)p * HDIM + col];
      else                            o = 0.0f;
      out[(size_t)p * HDIM + col] = o;
    }
  }
}

// ---------------------------------------------------------------------------
// Host launcher
// ---------------------------------------------------------------------------
extern "C" void kernel_launch(void* const* d_in, const int* in_sizes, int n_in,
                              void* d_out, int out_size, void* d_ws, size_t ws_size,
                              hipStream_t stream) {
  const float* x            = (const float*)d_in[0];
  const float* W_p          = (const float*)d_in[1];
  const float* W_pf         = (const float*)d_in[2];
  const float* z_Wih        = (const float*)d_in[3];
  const float* z_Whh        = (const float*)d_in[4];
  const float* z_Wf         = (const float*)d_in[5];
  const float* s_Wih        = (const float*)d_in[6];
  const float* s_Whh        = (const float*)d_in[7];
  const float* s_Wf         = (const float*)d_in[8];
  const int*   initial_map  = (const int*)d_in[9];
  const int*   order_matrix = (const int*)d_in[10];
  const int*   pooling      = (const int*)d_in[11];
  const int*   type_mask    = (const int*)d_in[12];

  const int M = in_sizes[11];          // pooling length
  const int P = out_size / HDIM;       // number of groups

  // Workspace layout (all 16B-aligned): 4 x [P][128] f32 acc, [P] i32, 8 bf16 weights
  float* accP = (float*)d_ws;
  float* accZ = accP + (size_t)P * HDIM;
  float* accS = accZ + (size_t)P * HDIM;
  float* acc0 = accS + (size_t)P * HDIM;
  int* typeAfter = (int*)(acc0 + (size_t)P * HDIM);
  unsigned short* wbf = (unsigned short*)(typeAfter + P);

  // 1) zero accumulators, type_after = -1
  long long nAcc = 4LL * P * HDIM;
  init_ws_kernel<<<2048, 256, 0, stream>>>(accP, nAcc, typeAfter, P);

  // 2) weights -> bf16 (order: Wp, Wpf, zWih, zWhh, zWf, sWih, sWhh, sWf)
  conv_weights_kernel<<<(WELEMS + 255) / 256, 256, 0, stream>>>(
      W_p, W_pf, z_Wih, z_Whh, z_Wf, s_Wih, s_Whh, s_Wf, wbf);

  // 3) type_after = segment_max(type_mask, pooling)
  type_after_kernel<<<(M + 255) / 256, 256, 0, stream>>>(pooling, type_mask, typeAfter, M);

  // 4) main fused gather + WMMA GEMMs + RNNs + masked segment sums
  tree_rnn_main<<<(M + 15) / 16, 32, 0, stream>>>(
      x, initial_map, order_matrix, pooling, type_mask, wbf,
      accP, accZ, accS, acc0, M);

  // 5) finalize per group (includes W_pf GEMM for P-type groups)
  finalize_kernel<<<(P + 15) / 16, 32, 0, stream>>>(
      accP, accZ, accS, acc0, typeAfter, wbf + 1 * WELEMS, (float*)d_out, P);
}
